// LinearAttention_16853451670184
// MI455X (gfx1250) — matrix operand order repaired and checked by reading, not compiled, not run on hardware
//
#include <hip/hip_runtime.h>

typedef __attribute__((ext_vector_type(16))) __bf16 v16bf;
typedef __attribute__((ext_vector_type(8)))  __bf16 v8bf;
typedef __attribute__((ext_vector_type(8)))  float  v8f;
typedef __attribute__((ext_vector_type(4)))  unsigned int u32x4;
typedef __attribute__((ext_vector_type(8)))  int i32x8;
typedef __attribute__((ext_vector_type(4)))  int i32x4;

#define B_    8
#define N_    4096
#define DIM_  1024
#define H_    16
#define HD_   64
#define MTOT  (B_ * N_)          // 32768 rows

#if __has_builtin(__builtin_amdgcn_tensor_load_to_lds)
#define USE_TDM 1
#endif

// ---------------------------------------------------------------------------
// WMMA helpers (CDNA5 16x16x32 bf16, f32 accumulate)
// ---------------------------------------------------------------------------
__device__ __forceinline__ v8f wmma_bf16(v16bf a, v16bf b, v8f c) {
  return __builtin_amdgcn_wmma_f32_16x16x32_bf16(
      false, a, false, b, (short)0, c, false, false);
}

// A fragment: 16x32 (MxK), tile row-major [m][k], ld in halves.
__device__ __forceinline__ v16bf frag_a(const __bf16* tile, int ld, int lane) {
  int row = lane & 15;
  int kb  = (lane < 16) ? 0 : 8;
  const __bf16* p = tile + row * ld + kb;
  v8bf lo = *(const v8bf*)(p);
  v8bf hi = *(const v8bf*)(p + 16);
  v16bf r;
#pragma unroll
  for (int i = 0; i < 8; ++i) { r[i] = lo[i]; r[i + 8] = hi[i]; }
  return r;
}

// B fragment: 32x16 (KxN), tile stored [n][k].
__device__ __forceinline__ v16bf frag_b(const __bf16* tile, int ld, int lane) {
  int col = lane & 15;
  int kb  = (lane < 16) ? 0 : 8;
  const __bf16* p = tile + col * ld + kb;
  v8bf lo = *(const v8bf*)(p);
  v8bf hi = *(const v8bf*)(p + 16);
  v16bf r;
#pragma unroll
  for (int i = 0; i < 8; ++i) { r[i] = lo[i]; r[i + 8] = hi[i]; }
  return r;
}

// ---------------------------------------------------------------------------
// TDM tile load: 2D tile (tileRows x rowLen bf16), row stride rowStrideElems,
// LDS rows padded +16B via D# pad (pad_interval code 4: 128B, pad_amount 3: 16B)
// ---------------------------------------------------------------------------
__device__ __forceinline__ void tdm_load_tile(const __bf16* gsrc, unsigned ldsAddr,
                                              int tileRows, int rowLen,
                                              int rowStrideElems,
                                              int padInterval, int padAmount) {
#ifdef USE_TDM
  unsigned long long ga = (unsigned long long)(size_t)gsrc;
  u32x4 g0;
  g0[0] = 1u;                                        // count=1 (valid)
  g0[1] = ldsAddr;                                   // lds_addr (bytes)
  g0[2] = (unsigned)(ga & 0xffffffffu);              // global_addr lo
  g0[3] = (unsigned)((ga >> 32) & 0x01ffffffu) | (2u << 30);  // hi | type=2
  i32x8 g1;
  g1[0] = (1 << 16) | (1 << 20) | (padInterval << 22) | (padAmount << 25);
  g1[1] = (int)(((unsigned)rowLen & 0xffffu) << 16);           // tensor_dim0 lo16
  g1[2] = (int)((((unsigned)rowLen >> 16) & 0xffffu) |
                (((unsigned)tileRows & 0xffffu) << 16));       // dim0 hi | dim1 lo
  g1[3] = (int)(((unsigned)tileRows >> 16) |
                (((unsigned)rowLen & 0xffffu) << 16));         // dim1 hi | tile_dim0
  g1[4] = (int)((unsigned)tileRows & 0xffffu);                 // tile_dim1
  g1[5] = rowStrideElems;                                      // dim0_stride lo32
  g1[6] = 0;
  g1[7] = 0;
  i32x4 gz = {0, 0, 0, 0};
  i32x8 gz8 = {0, 0, 0, 0, 0, 0, 0, 0};
  __builtin_amdgcn_tensor_load_to_lds(g0, g1, gz, gz, gz8, 0);
#else
  (void)gsrc; (void)ldsAddr; (void)tileRows; (void)rowLen;
  (void)rowStrideElems; (void)padInterval; (void)padAmount;
#endif
}

// ---------------------------------------------------------------------------
// f32 -> bf16 cast (8-element chunks)
// ---------------------------------------------------------------------------
__global__ __launch_bounds__(256)
void cast_f32_bf16(const float* __restrict__ src, __bf16* __restrict__ dst, int n) {
  int i = (blockIdx.x * 256 + threadIdx.x) * 8;
  if (i >= n) return;
  float4 a = *(const float4*)(src + i);
  float4 b = *(const float4*)(src + i + 4);
  v8bf o;
  o[0] = (__bf16)a.x; o[1] = (__bf16)a.y; o[2] = (__bf16)a.z; o[3] = (__bf16)a.w;
  o[4] = (__bf16)b.x; o[5] = (__bf16)b.y; o[6] = (__bf16)b.z; o[7] = (__bf16)b.w;
  *(v8bf*)(dst + i) = o;
}

// ---------------------------------------------------------------------------
// Projection GEMM: Y(M x Nout) = X(M x K) @ W(Nout x K)^T + bias   (bf16 in)
// 256 threads (8 wave32), 128x256 C block tile, 64x64 wave tiles, K-step 64,
// double-buffered LDS, tiles staged by TDM (wave 0 issues, s_wait_tensorcnt).
// ---------------------------------------------------------------------------
template <bool OUT_BF16>
__global__ __launch_bounds__(256)
void proj_gemm(const __bf16* __restrict__ X, const __bf16* __restrict__ W,
               const float* __restrict__ bias, void* __restrict__ Yv,
               int M, int K, int Nout) {
  __shared__ __bf16 As[2][128][72];   // [buf][m][k], 144B row stride (36.9 KB)
  __shared__ __bf16 Bs[2][256][72];   // [buf][n][k]                  (73.7 KB)

  const int t    = threadIdx.x;
  const int lane = t & 31;
  const int w    = t >> 5;
  const int m0   = blockIdx.y * 128;
  const int n0   = blockIdx.x * 256;
  const int wm   = (w >> 2) * 64;          // wave tile: 64 x 64
  const int wn   = (w & 3) * 64;

  v8f acc[4][4] = {};

#ifdef USE_TDM
  if (w == 0) {
    tdm_load_tile(X + (size_t)m0 * K, (unsigned)(size_t)&As[0][0][0], 128, 64, K, 4, 3);
    tdm_load_tile(W + (size_t)n0 * K, (unsigned)(size_t)&Bs[0][0][0], 256, 64, K, 4, 3);
  }
#endif

  for (int kk = 0, buf = 0; kk < K; kk += 64, buf ^= 1) {
#ifdef USE_TDM
    if (w == 0) {
      if (kk + 64 < K) {
        tdm_load_tile(X + (size_t)m0 * K + kk + 64,
                      (unsigned)(size_t)&As[buf ^ 1][0][0], 128, 64, K, 4, 3);
        tdm_load_tile(W + (size_t)n0 * K + kk + 64,
                      (unsigned)(size_t)&Bs[buf ^ 1][0][0], 256, 64, K, 4, 3);
        __builtin_amdgcn_s_wait_tensorcnt((short)2);
      } else {
        __builtin_amdgcn_s_wait_tensorcnt((short)0);
      }
    }
#else
    {
      // fallback staging: pure bf16 vector copy
      const int row = t >> 1;
      const int c   = (t & 1) * 32;
      const v8bf* px = (const v8bf*)(X + (size_t)(m0 + row) * K + kk + c);
#pragma unroll
      for (int j = 0; j < 4; ++j) *(v8bf*)&As[buf][row][c + 8 * j] = px[j];
      const v8bf* pw = (const v8bf*)(W + (size_t)(n0 + t) * K + kk);
#pragma unroll
      for (int j = 0; j < 8; ++j) *(v8bf*)&Bs[buf][t][8 * j] = pw[j];
    }
#endif
    __syncthreads();

#pragma unroll
    for (int kc = 0; kc < 2; ++kc) {
      v16bf af[4];
#pragma unroll
      for (int mi = 0; mi < 4; ++mi)
        af[mi] = frag_a(&As[buf][wm + mi * 16][kc * 32], 72, lane);
#pragma unroll
      for (int ni = 0; ni < 4; ++ni) {
        const v16bf bfv = frag_b(&Bs[buf][wn + ni * 16][kc * 32], 72, lane);
#pragma unroll
        for (int mi = 0; mi < 4; ++mi)
          acc[mi][ni] = wmma_bf16(af[mi], bfv, acc[mi][ni]);
      }
    }
    __syncthreads();
  }

#pragma unroll
  for (int ni = 0; ni < 4; ++ni) {
    const int col = n0 + wn + ni * 16 + (lane & 15);
    const float bv = bias[col];
#pragma unroll
    for (int mi = 0; mi < 4; ++mi) {
#pragma unroll
      for (int r = 0; r < 8; ++r) {
        const int m = m0 + wm + mi * 16 + r + ((lane >= 16) ? 8 : 0);
        const float vv = acc[mi][ni][r] + bv;
        if (OUT_BF16) ((__bf16*)Yv)[(size_t)m * Nout + col] = (__bf16)vv;
        else          ((float*)Yv)[(size_t)m * Nout + col]  = vv;
      }
    }
  }
}

// ---------------------------------------------------------------------------
// q column-softmax stats (softmax over n, axis=1): per (b,d) max & 1/sumexp
// ---------------------------------------------------------------------------
__global__ __launch_bounds__(256)
void colstats_q(const float* __restrict__ Q, float* __restrict__ qmax,
                float* __restrict__ qrsum) {
  const int d = blockIdx.x * 256 + threadIdx.x;
  const int b = blockIdx.y;
  const float* p = Q + (size_t)b * N_ * DIM_ + d;
  float m = -3.4e38f;
  for (int n = 0; n < N_; ++n) m = fmaxf(m, p[(size_t)n * DIM_]);
  float s = 0.f;
  for (int n = 0; n < N_; ++n) s += __expf(p[(size_t)n * DIM_] - m);
  qmax[b * DIM_ + d]  = m;
  qrsum[b * DIM_ + d] = 1.0f / s;
}

// ---------------------------------------------------------------------------
// k row-softmax stats (softmax over features, axis=2): per (b,n) max & 1/sumexp
// ---------------------------------------------------------------------------
__global__ __launch_bounds__(256)
void rowstats_k(const float* __restrict__ Kf, float* __restrict__ kmax,
                float* __restrict__ krsum) {
  __shared__ float red[256];
  const int row = blockIdx.x;
  const int t = threadIdx.x;
  const float* p = Kf + (size_t)row * DIM_;
  float m = -3.4e38f;
  for (int i = t; i < DIM_; i += 256) m = fmaxf(m, p[i]);
  red[t] = m; __syncthreads();
  for (int s = 128; s > 0; s >>= 1) {
    if (t < s) red[t] = fmaxf(red[t], red[t + s]);
    __syncthreads();
  }
  m = red[0]; __syncthreads();
  float sum = 0.f;
  for (int i = t; i < DIM_; i += 256) sum += __expf(p[i] - m);
  red[t] = sum; __syncthreads();
  for (int s = 128; s > 0; s >>= 1) {
    if (t < s) red[t] += red[t + s];
    __syncthreads();
  }
  if (t == 0) { kmax[row] = m; krsum[row] = 1.0f / red[0]; }
}

// ---------------------------------------------------------------------------
// kv = softmax(k)^T @ v per (b,h):  64 x 64, K = 4096 (over n)
// ---------------------------------------------------------------------------
__global__ __launch_bounds__(256)
void kv_gemm(const float* __restrict__ Kf, const __bf16* __restrict__ Vb,
             const float* __restrict__ kmax, const float* __restrict__ krsum,
             const int* __restrict__ mask, __bf16* __restrict__ KV) {
  __shared__ __bf16 Ak[64][40];
  __shared__ __bf16 Bv[64][40];

  const int bh = blockIdx.x;
  const int b  = bh >> 4;
  const int h  = bh & 15;
  const int t = threadIdx.x, lane = t & 31, w = t >> 5;
  const int mt    = (w & 3) * 16;
  const int npair = (w >> 2) * 2;
  v8f acc[2] = {};

  const int nl = t >> 3;
  const int f8 = (t & 7) * 8;

  for (int n0 = 0; n0 < N_; n0 += 32) {
    const int rowIdx = b * N_ + n0 + nl;
    const float rmax = kmax[rowIdx];
    const float rrs  = krsum[rowIdx];
    const int   mk   = mask[rowIdx];
    const float4* kp = (const float4*)(Kf + (size_t)rowIdx * DIM_ + h * HD_ + f8);
    float4 f0 = kp[0], f1 = kp[1];
    const float vals[8] = {f0.x, f0.y, f0.z, f0.w, f1.x, f1.y, f1.z, f1.w};
#pragma unroll
    for (int j = 0; j < 8; ++j) {
      const float nv = mk ? (1.0f / 1024.0f) : __expf(vals[j] - rmax) * rrs;
      Ak[f8 + j][nl] = (__bf16)nv;
    }
    const v8bf vv = *(const v8bf*)(Vb + (size_t)rowIdx * DIM_ + h * HD_ + f8);
#pragma unroll
    for (int j = 0; j < 8; ++j) Bv[f8 + j][nl] = vv[j];
    __syncthreads();

    const v16bf af = frag_a(&Ak[mt][0], 40, lane);
    const v16bf b0 = frag_b(&Bv[npair * 16][0], 40, lane);
    const v16bf b1 = frag_b(&Bv[(npair + 1) * 16][0], 40, lane);
    acc[0] = wmma_bf16(af, b0, acc[0]);
    acc[1] = wmma_bf16(af, b1, acc[1]);
    __syncthreads();
  }

#pragma unroll
  for (int ni = 0; ni < 2; ++ni) {
    const int e = (npair + ni) * 16 + (lane & 15);
#pragma unroll
    for (int r = 0; r < 8; ++r) {
      const int d = mt + r + ((lane >= 16) ? 8 : 0);
      KV[((size_t)bh * HD_ + d) * HD_ + e] = (__bf16)acc[ni][r];
    }
  }
}

// ---------------------------------------------------------------------------
// out = softmax_n(q) @ kv per head: 128 rows x 64, K = 64
// ---------------------------------------------------------------------------
__global__ __launch_bounds__(256)
void out_gemm(const float* __restrict__ Qf, const __bf16* __restrict__ KV,
              const float* __restrict__ qmax, const float* __restrict__ qrsum,
              __bf16* __restrict__ Attn) {
  __shared__ __bf16 Aq[128][72];
  __shared__ __bf16 Bs[64][72];

  const int h    = blockIdx.y;
  const int m0   = blockIdx.x * 128;
  const int b    = m0 >> 12;
  const int t = threadIdx.x, lane = t & 31, w = t >> 5;

  {
    const int row = t >> 1;
    const int fb  = (t & 1) * 32;
    const float* qp = Qf + (size_t)(m0 + row) * DIM_ + h * HD_ + fb;
#pragma unroll
    for (int j = 0; j < 32; j += 4) {
      float4 f = *(const float4*)(qp + j);
      const float vals[4] = {f.x, f.y, f.z, f.w};
#pragma unroll
      for (int u = 0; u < 4; ++u) {
        const int dcol = h * HD_ + fb + j + u;
        Aq[row][fb + j + u] =
            (__bf16)(__expf(vals[u] - qmax[b * DIM_ + dcol]) * qrsum[b * DIM_ + dcol]);
      }
    }
  }
  if (t < 64) {
    const int d = t;
    const v8bf* kp = (const v8bf*)(KV + ((size_t)(b * H_ + h) * HD_ + d) * HD_);
#pragma unroll
    for (int jj = 0; jj < 8; ++jj) {
      v8bf kvv = kp[jj];
#pragma unroll
      for (int u = 0; u < 8; ++u) Bs[jj * 8 + u][d] = kvv[u];
    }
  }
  __syncthreads();

  const int mrow = w * 16;
  v8f acc[4] = {};
#pragma unroll
  for (int ks = 0; ks < 2; ++ks) {
    const v16bf af = frag_a(&Aq[mrow][ks * 32], 72, lane);
#pragma unroll
    for (int ni = 0; ni < 4; ++ni) {
      const v16bf bfr = frag_b(&Bs[ni * 16][ks * 32], 72, lane);
      acc[ni] = wmma_bf16(af, bfr, acc[ni]);
    }
  }

#pragma unroll
  for (int ni = 0; ni < 4; ++ni) {
    const int e = ni * 16 + (lane & 15);
#pragma unroll
    for (int r = 0; r < 8; ++r) {
      const int row = mrow + r + ((lane >= 16) ? 8 : 0);
      Attn[(size_t)(m0 + row) * DIM_ + h * HD_ + e] = (__bf16)acc[ni][r];
    }
  }
}

// ---------------------------------------------------------------------------
// Launch
// ---------------------------------------------------------------------------
extern "C" void kernel_launch(void* const* d_in, const int* in_sizes, int n_in,
                              void* d_out, int out_size, void* d_ws, size_t ws_size,
                              hipStream_t stream) {
  const float* query = (const float*)d_in[0];
  const float* key   = (const float*)d_in[1];
  const float* value = (const float*)d_in[2];
  const int*   mask  = (const int*)d_in[3];
  const float* Wq = (const float*)d_in[4];  const float* bq = (const float*)d_in[5];
  const float* Wk = (const float*)d_in[6];  const float* bk = (const float*)d_in[7];
  const float* Wv = (const float*)d_in[8];  const float* bv = (const float*)d_in[9];
  const float* Wo = (const float*)d_in[10]; const float* bo = (const float*)d_in[11];

  char* ws = (char*)d_ws;
  size_t off = 0;
  const size_t actE = (size_t)MTOT * DIM_;          // 33.5M elements
  const size_t wE   = (size_t)DIM_ * DIM_;          // 1M elements
  __bf16* xq_bf = (__bf16*)(ws + off); off += actE * 2;
  __bf16* xk_bf = (__bf16*)(ws + off); off += actE * 2;
  __bf16* xv_bf = (__bf16*)(ws + off); off += actE * 2;
  __bf16* wq_bf = (__bf16*)(ws + off); off += wE * 2;
  __bf16* wk_bf = (__bf16*)(ws + off); off += wE * 2;
  __bf16* wv_bf = (__bf16*)(ws + off); off += wE * 2;
  __bf16* wo_bf = (__bf16*)(ws + off); off += wE * 2;
  float*  q_f32 = (float*)(ws + off);  off += actE * 4;
  float*  k_f32 = (float*)(ws + off);  off += actE * 4;
  __bf16* v_bf  = (__bf16*)(ws + off); off += actE * 2;
  __bf16* attn  = (__bf16*)(ws + off); off += actE * 2;
  __bf16* kv_bf = (__bf16*)(ws + off); off += (size_t)B_ * H_ * HD_ * HD_ * 2;
  float*  qmaxp = (float*)(ws + off);  off += (size_t)B_ * DIM_ * 4;
  float*  qrsum = (float*)(ws + off);  off += (size_t)B_ * DIM_ * 4;
  float*  kmaxp = (float*)(ws + off);  off += (size_t)MTOT * 4;
  float*  krsum = (float*)(ws + off);  off += (size_t)MTOT * 4;

  // pre-cast activations & weights to bf16
  const int actBlocks = (int)(actE / 8 / 256);
  const int wBlocks   = (int)(wE / 8 / 256);
  cast_f32_bf16<<<actBlocks, 256, 0, stream>>>(query, xq_bf, (int)actE);
  cast_f32_bf16<<<actBlocks, 256, 0, stream>>>(key,   xk_bf, (int)actE);
  cast_f32_bf16<<<actBlocks, 256, 0, stream>>>(value, xv_bf, (int)actE);
  cast_f32_bf16<<<wBlocks, 256, 0, stream>>>(Wq, wq_bf, (int)wE);
  cast_f32_bf16<<<wBlocks, 256, 0, stream>>>(Wk, wk_bf, (int)wE);
  cast_f32_bf16<<<wBlocks, 256, 0, stream>>>(Wv, wv_bf, (int)wE);
  cast_f32_bf16<<<wBlocks, 256, 0, stream>>>(Wo, wo_bf, (int)wE);

  const dim3 gproj(DIM_ / 256, MTOT / 128);   // (4, 256)

  proj_gemm<false><<<gproj, 256, 0, stream>>>(xq_bf, wq_bf, bq, q_f32, MTOT, DIM_, DIM_);
  proj_gemm<false><<<gproj, 256, 0, stream>>>(xk_bf, wk_bf, bk, k_f32, MTOT, DIM_, DIM_);
  proj_gemm<true ><<<gproj, 256, 0, stream>>>(xv_bf, wv_bf, bv, v_bf,  MTOT, DIM_, DIM_);

  colstats_q<<<dim3(DIM_ / 256, B_), 256, 0, stream>>>(q_f32, qmaxp, qrsum);
  rowstats_k<<<MTOT, 256, 0, stream>>>(k_f32, kmaxp, krsum);

  kv_gemm<<<B_ * H_, 256, 0, stream>>>(k_f32, v_bf, kmaxp, krsum, mask, kv_bf);
  out_gemm<<<dim3(MTOT / 128, H_), 256, 0, stream>>>(q_f32, kv_bf, qmaxp, qrsum, attn);

  proj_gemm<false><<<gproj, 256, 0, stream>>>(attn, wo_bf, bo, (float*)d_out, MTOT, DIM_, DIM_);
}
